// CrossAttentionNetwork_27900107555210
// MI455X (gfx1250) — matrix operand order repaired and checked by reading, not compiled, not run on hardware
//
#include <hip/hip_runtime.h>
#include <hip/hip_bf16.h>
#include <math.h>

// Problem constants (from reference)
#define B_    16
#define NX_   512
#define LY_   1024
#define DIN_  768
#define DK_   512     // == DV
#define NH_   8
#define HD_   64      // head dim
#define SCALE_ 0.125f // 1/sqrt(64); power of two -> exact when folded into f16 Q

typedef __attribute__((ext_vector_type(16))) _Float16 v16h;
typedef __attribute__((ext_vector_type(8)))  float    v8f;

union Frag16 {
    v16h     v;
    _Float16 h[16];
    uint4    q[2];   // two 16B loads fill 16 f16
};

static __device__ inline v8f wmma_f16(const Frag16& a, const Frag16& b, v8f c) {
    // D = A(16x32 f16) * B(32x16 f16) + C(16x16 f32)
    return __builtin_amdgcn_wmma_f32_16x16x32_f16(
        /*neg_a=*/false, a.v, /*neg_b=*/false, b.v,
        /*c_mod=*/(short)0, c, /*reuse_a=*/false, /*reuse_b=*/false);
}

// ---------------------------------------------------------------------------
// f32 -> f16 bulk conversion (memory-bound, one pass). n4 = count/4.
// ---------------------------------------------------------------------------
__global__ __launch_bounds__(256)
void cvt_kernel(const float* __restrict__ in, _Float16* __restrict__ outp, int n4)
{
    const int i = blockIdx.x * blockDim.x + threadIdx.x;
    if (i < n4) {
        const float4 f = ((const float4*)in)[i];
        union { _Float16 h[4]; uint2 u; } o;
        o.h[0] = (_Float16)f.x;
        o.h[1] = (_Float16)f.y;
        o.h[2] = (_Float16)f.z;
        o.h[3] = (_Float16)f.w;
        ((uint2*)outp)[i] = o.u;
    }
}

// ---------------------------------------------------------------------------
// Projection GEMM on pre-converted f16 operands:
//   out[m,n] = sum_d X[m,d] * W[n,d]      (W stored (out,in))
// One wave computes a 32x64 output tile as 2x4 WMMA accumulators:
// per k-step 12 b128 loads feed 8 WMMAs.
// MODE 0: store f32 row-major (Q)           -> outF
// MODE 1: store f16 row-major (K)           -> outH
// MODE 2: store f16 transposed per head (V) -> outH as Vt[b][h][d][LY]
// ---------------------------------------------------------------------------
template<int MODE>
static __device__ inline void store_tile(const v8f& c, int mbase, int nbase,
                                         int half, int lr,
                                         float* __restrict__ outF,
                                         _Float16* __restrict__ outH)
{
    #pragma unroll
    for (int r = 0; r < 8; ++r) {
        const int mg = mbase + r + 8 * half;
        const int ng = nbase + lr;
        if (MODE == 0) {
            outF[(size_t)mg * DK_ + ng] = c[r];
        } else if (MODE == 1) {
            outH[(size_t)mg * DK_ + ng] = (_Float16)c[r];
        } else { // Vt[b][h][d][LY]
            const int bb = mg / LY_;
            const int ml = mg - bb * LY_;
            const int hh = ng >> 6;
            const int d  = ng & 63;
            outH[(((size_t)bb * NH_ + hh) * HD_ + d) * LY_ + ml] = (_Float16)c[r];
        }
    }
}

template<int MODE>
__global__ __launch_bounds__(32)
void proj_kernel(const _Float16* __restrict__ Xh, const _Float16* __restrict__ Wh,
                 float* __restrict__ outF, _Float16* __restrict__ outH)
{
    const int lane = threadIdx.x & 31;
    const int half = lane >> 4;   // 0 or 1
    const int lr   = lane & 15;
    const int tn   = blockIdx.x;  // 64-col tile over DK_
    const int tm   = blockIdx.y;  // 32-row tile over flattened B*seq

    const _Float16* x0 = Xh + (size_t)(tm * 32 + lr) * DIN_;
    const _Float16* x1 = x0 + (size_t)16 * DIN_;
    const _Float16* w0 = Wh + (size_t)(tn * 64 + lr) * DIN_;
    const _Float16* w1 = w0 + (size_t)16 * DIN_;
    const _Float16* w2 = w0 + (size_t)32 * DIN_;
    const _Float16* w3 = w0 + (size_t)48 * DIN_;

    v8f c00 = {}, c01 = {}, c02 = {}, c03 = {};
    v8f c10 = {}, c11 = {}, c12 = {}, c13 = {};
    for (int kb = 0; kb < DIN_; kb += 32) {
        Frag16 a0, a1, b0, b1, b2, b3;
        a0.q[0] = *(const uint4*)(x0 + kb + half * 8);
        a0.q[1] = *(const uint4*)(x0 + kb + 16 + half * 8);
        a1.q[0] = *(const uint4*)(x1 + kb + half * 8);
        a1.q[1] = *(const uint4*)(x1 + kb + 16 + half * 8);
        b0.q[0] = *(const uint4*)(w0 + kb + half * 8);
        b0.q[1] = *(const uint4*)(w0 + kb + 16 + half * 8);
        b1.q[0] = *(const uint4*)(w1 + kb + half * 8);
        b1.q[1] = *(const uint4*)(w1 + kb + 16 + half * 8);
        b2.q[0] = *(const uint4*)(w2 + kb + half * 8);
        b2.q[1] = *(const uint4*)(w2 + kb + 16 + half * 8);
        b3.q[0] = *(const uint4*)(w3 + kb + half * 8);
        b3.q[1] = *(const uint4*)(w3 + kb + 16 + half * 8);
        c00 = wmma_f16(a0, b0, c00);
        c01 = wmma_f16(a0, b1, c01);
        c02 = wmma_f16(a0, b2, c02);
        c03 = wmma_f16(a0, b3, c03);
        c10 = wmma_f16(a1, b0, c10);
        c11 = wmma_f16(a1, b1, c11);
        c12 = wmma_f16(a1, b2, c12);
        c13 = wmma_f16(a1, b3, c13);
    }
    store_tile<MODE>(c00, tm * 32,      tn * 64,      half, lr, outF, outH);
    store_tile<MODE>(c01, tm * 32,      tn * 64 + 16, half, lr, outF, outH);
    store_tile<MODE>(c02, tm * 32,      tn * 64 + 32, half, lr, outF, outH);
    store_tile<MODE>(c03, tm * 32,      tn * 64 + 48, half, lr, outF, outH);
    store_tile<MODE>(c10, tm * 32 + 16, tn * 64,      half, lr, outF, outH);
    store_tile<MODE>(c11, tm * 32 + 16, tn * 64 + 16, half, lr, outF, outH);
    store_tile<MODE>(c12, tm * 32 + 16, tn * 64 + 32, half, lr, outF, outH);
    store_tile<MODE>(c13, tm * 32 + 16, tn * 64 + 48, half, lr, outF, outH);
}

// ---------------------------------------------------------------------------
// Attention: one block = 16 query rows x one head x one batch. 4 waves.
//  Phase A: S(16x1024) = (Q*scale)(16x64) * K^T -> f32 LDS (padded rows);
//           scale folded into f16 Q fragments (power of two => exact),
//           so the WMMA accumulator is stored with no post-multiply.
//  Phase B: p = softmax(S); out = softmax(1-p) == softmax(-p) -> f16 LDS
//  Phase C: each wave: one 16x16 col-tile of O = P(16x1024) * V(1024x64),
//           A-fragments straight from f16 LDS (ds_load_b128), + residual q.
// ---------------------------------------------------------------------------
#define SROW 1032  // f32 score row stride (1024 + 8 pad)
#define PROW 1032  // f16 prob  row stride (2064B, 16B-aligned rows)

__global__ __launch_bounds__(128)
void attn_kernel(const float* __restrict__ Qf, const _Float16* __restrict__ Kh,
                 const _Float16* __restrict__ Vt, float* __restrict__ out)
{
    __shared__ float    s[16 * SROW];
    __shared__ _Float16 ph[16 * PROW];

    const int lane = threadIdx.x & 31;
    const int wid  = threadIdx.x >> 5;   // 0..3
    const int half = lane >> 4;
    const int lr   = lane & 15;
    const int qt   = blockIdx.x;         // 0..31 (NX/16)
    const int h    = blockIdx.y;         // 0..7
    const int b    = blockIdx.z;         // 0..15

    // Q A-fragments, scale pre-folded (exact: 0.125 is a power of two)
    Frag16 qa0, qa1;
    {
        const float* qrow = Qf + ((size_t)b * NX_ + qt * 16 + lr) * DK_ + h * HD_;
        #pragma unroll
        for (int i = 0; i < 8; ++i) {
            qa0.h[i]     = (_Float16)(qrow[half * 8 + i]      * SCALE_);
            qa0.h[8 + i] = (_Float16)(qrow[16 + half * 8 + i] * SCALE_);
            qa1.h[i]     = (_Float16)(qrow[32 + half * 8 + i] * SCALE_);
            qa1.h[8 + i] = (_Float16)(qrow[48 + half * 8 + i] * SCALE_);
        }
    }

    // ---- Phase A: scores into f32 LDS ----
    for (int lt = wid; lt < LY_ / 16; lt += 4) {
        const _Float16* krow =
            Kh + ((size_t)b * LY_ + lt * 16 + lr) * DK_ + h * HD_ + half * 8;
        Frag16 b0, b1;
        b0.q[0] = *(const uint4*)(krow);
        b0.q[1] = *(const uint4*)(krow + 16);
        b1.q[0] = *(const uint4*)(krow + 32);
        b1.q[1] = *(const uint4*)(krow + 48);
        v8f c = {};
        c = wmma_f16(qa0, b0, c);
        c = wmma_f16(qa1, b1, c);
        #pragma unroll
        for (int r = 0; r < 8; ++r)
            s[(r + 8 * half) * SROW + lt * 16 + lr] = c[r];
    }
    __syncthreads();

    // ---- Phase B: double softmax, 4 rows per wave, result in f16 LDS ----
    for (int rr = 0; rr < 4; ++rr) {
        const int rowi = wid * 4 + rr;
        float*     row  = s  + rowi * SROW;
        _Float16*  prow = ph + rowi * PROW;
        float mx = -1e30f;
        for (int j = lane; j < LY_; j += 32) mx = fmaxf(mx, row[j]);
        #pragma unroll
        for (int o = 16; o > 0; o >>= 1) mx = fmaxf(mx, __shfl_xor(mx, o, 32));
        float sum = 0.f;
        for (int j = lane; j < LY_; j += 32) sum += __expf(row[j] - mx);
        #pragma unroll
        for (int o = 16; o > 0; o >>= 1) sum += __shfl_xor(sum, o, 32);
        const float inv = 1.0f / sum;
        float mn = 1e30f;
        for (int j = lane; j < LY_; j += 32) {
            float p = __expf(row[j] - mx) * inv;
            row[j] = p;
            mn = fminf(mn, p);
        }
        #pragma unroll
        for (int o = 16; o > 0; o >>= 1) mn = fminf(mn, __shfl_xor(mn, o, 32));
        float sum2 = 0.f;
        for (int j = lane; j < LY_; j += 32) {
            float t = __expf(mn - row[j]);   // softmax(1-p) == softmax(-p)
            row[j] = t;
            sum2 += t;
        }
        #pragma unroll
        for (int o = 16; o > 0; o >>= 1) sum2 += __shfl_xor(sum2, o, 32);
        const float inv2 = 1.0f / sum2;
        for (int j = lane; j < LY_; j += 32)
            prow[j] = (_Float16)(row[j] * inv2);
    }
    __syncthreads();

    // ---- Phase C: O = P * V for column tile n0 = wid*16, + residual ----
    const int n0 = wid * 16;
    const _Float16* vrow =
        Vt + (((size_t)b * NH_ + h) * HD_ + n0 + lr) * LY_ + half * 8;
    const _Float16* pr = ph + lr * PROW;
    v8f c = {};
    for (int kb = 0; kb < LY_; kb += 32) {
        Frag16 a, bb;
        a.q[0]  = *(const uint4*)(pr + kb + half * 8);
        a.q[1]  = *(const uint4*)(pr + kb + 16 + half * 8);
        bb.q[0] = *(const uint4*)(vrow + kb);
        bb.q[1] = *(const uint4*)(vrow + kb + 16);
        c = wmma_f16(a, bb, c);
    }
    #pragma unroll
    for (int r = 0; r < 8; ++r) {
        const int mg = qt * 16 + r + 8 * half;
        const size_t oidx = ((size_t)b * NX_ + mg) * DK_ + h * HD_ + n0 + lr;
        out[oidx] = Qf[oidx] + c[r];   // residual add of q
    }
}

extern "C" void kernel_launch(void* const* d_in, const int* in_sizes, int n_in,
                              void* d_out, int out_size, void* d_ws, size_t ws_size,
                              hipStream_t stream) {
    (void)in_sizes; (void)n_in; (void)out_size; (void)ws_size;
    const float* x  = (const float*)d_in[0];
    const float* y  = (const float*)d_in[1];
    const float* Wq = (const float*)d_in[2];
    const float* Wk = (const float*)d_in[3];
    const float* Wv = (const float*)d_in[4];
    float* out = (float*)d_out;

    // Workspace layout:
    //   Qf f32 (16MB) | Kh f16 (16MB) | Vt f16 (16MB) |
    //   Xh f16 (12MB) | Yh f16 (24MB) | Wqh/Wkh/Wvh f16 (0.75MB each)
    const size_t nQ  = (size_t)B_ * NX_ * DK_;
    const size_t nKV = (size_t)B_ * LY_ * DK_;
    const size_t nX  = (size_t)B_ * NX_ * DIN_;
    const size_t nY  = (size_t)B_ * LY_ * DIN_;
    const size_t nW  = (size_t)DK_ * DIN_;

    float*    Qf  = (float*)d_ws;
    _Float16* Kh  = (_Float16*)(Qf + nQ);
    _Float16* Vt  = Kh + nKV;
    _Float16* Xh  = Vt + nKV;
    _Float16* Yh  = Xh + nX;
    _Float16* Wqh = Yh + nY;
    _Float16* Wkh = Wqh + nW;
    _Float16* Wvh = Wkh + nW;

    // Bulk f32 -> f16 conversions (all counts divisible by 4)
    cvt_kernel<<<(int)((nX / 4 + 255) / 256), 256, 0, stream>>>(x,  Xh,  (int)(nX / 4));
    cvt_kernel<<<(int)((nY / 4 + 255) / 256), 256, 0, stream>>>(y,  Yh,  (int)(nY / 4));
    cvt_kernel<<<(int)((nW / 4 + 255) / 256), 256, 0, stream>>>(Wq, Wqh, (int)(nW / 4));
    cvt_kernel<<<(int)((nW / 4 + 255) / 256), 256, 0, stream>>>(Wk, Wkh, (int)(nW / 4));
    cvt_kernel<<<(int)((nW / 4 + 255) / 256), 256, 0, stream>>>(Wv, Wvh, (int)(nW / 4));

    // Projections: 32x64 tile per wave (2x4 WMMA accumulators), f16 operands
    proj_kernel<0><<<dim3(DK_ / 64, (B_ * NX_) / 32), 32, 0, stream>>>(Xh, Wqh, Qf, nullptr);
    proj_kernel<1><<<dim3(DK_ / 64, (B_ * LY_) / 32), 32, 0, stream>>>(Yh, Wkh, nullptr, Kh);
    proj_kernel<2><<<dim3(DK_ / 64, (B_ * LY_) / 32), 32, 0, stream>>>(Yh, Wvh, nullptr, Vt);

    // Attention
    attn_kernel<<<dim3(NX_ / 16, NH_, B_), 128, 0, stream>>>(Qf, Kh, Vt, out);
}
